// PureSparseBackboneCol_Ex1_38079180046732
// MI455X (gfx1250) — matrix operand order repaired and checked by reading, not compiled, not run on hardware
//
#include <hip/hip_runtime.h>
#include <hip/hip_bf16.h>

typedef __attribute__((ext_vector_type(16))) __bf16 v16bf;
typedef __attribute__((ext_vector_type(8)))  float  v8f;
typedef int v4i_vs __attribute__((__vector_size__(16)));
typedef __attribute__((address_space(1))) v4i_vs* gptr_v4i;
typedef __attribute__((address_space(3))) v4i_vs* lptr_v4i;

union Frag16 { unsigned int u32[8]; uint4 u128[2]; v16bf v; };

#if __has_builtin(__builtin_amdgcn_global_load_async_to_lds_b128)
#define USE_ASYNC_LDS 1
#else
#define USE_ASYNC_LDS 0
#endif

__device__ __forceinline__ void async_b128(const float* src, float* ldst) {
#if USE_ASYNC_LDS
  __builtin_amdgcn_global_load_async_to_lds_b128((gptr_v4i)src, (lptr_v4i)ldst, 0, 0);
#else
  *(float4*)ldst = *(const float4*)src;
#endif
}

template<int N>
__device__ __forceinline__ void wait_asynccnt() {
#if USE_ASYNC_LDS
#if __has_builtin(__builtin_amdgcn_s_wait_asynccnt)
  __builtin_amdgcn_s_wait_asynccnt(N);
#else
  asm volatile("s_wait_asynccnt %0" :: "i"(N) : "memory");
#endif
#endif
}

__device__ __forceinline__ unsigned int pack_bf16_pair(float lo, float hi) {
  float2 f2; f2.x = lo; f2.y = hi;
  __hip_bfloat162 hv = __float22bfloat162_rn(f2);   // v_cvt_pk_bf16_f32 on gfx1250
  unsigned int u; __builtin_memcpy(&u, &hv, 4);
  return u;
}

// Stage one 32-row x 32-channel fp32 tile into LDS via async DMA (2 B128/thread).
// Invalid lanes (missing neighbor / row tail / channel pad) fetch from a zeros line,
// so every wave issues exactly 2 async ops -> wave-uniform ASYNCcnt.
__device__ __forceinline__ void stage_tile(const float* __restrict__ x,
                                           const int* __restrict__ map,
                                           const float* __restrict__ zeros,
                                           float (*Xs)[36],
                                           int tid, int row0, int M, int K, int Cin,
                                           int k, int cc) {
  #pragma unroll
  for (int half = 0; half < 2; ++half) {
    const int u    = tid + half * 128;
    const int r    = u >> 3;            // 0..31
    const int qq   = u & 7;             // 16B chunk
    const int grow = row0 + r;
    const int idx  = (grow < M) ? map[(size_t)grow * K + k] : -1;
    const int c0   = (cc << 5) + (qq << 2);
    const float* src = (idx >= 0 && (c0 + 3) < Cin)
                     ? (x + (size_t)idx * Cin + c0) : zeros;
    async_b128(src, &Xs[r][qq << 2]);
  }
}

// -------------------------------------------------------------------------
// Sparse conv: y[M,Cout] = sum_k gather(x, map[:,k]) @ W[k]  (bf16 WMMA, f32 acc)
// Block = 128 threads (4 waves) = 32 output rows. Wave -> (row-half, col-tile):
// rsel = wave>>1 picks rows, ntb = wave&1 picks tile parity; TPW tiles per wave.
// Cout = 32*TPW. Double-buffered async-LDS gather overlaps DMA with WMMA.
// -------------------------------------------------------------------------
template<int TPW>
__launch_bounds__(128)
__global__ void sparse_conv_wmma(const float* __restrict__ x,
                                 const unsigned int* __restrict__ Wp,
                                 const int*   __restrict__ map,
                                 const float* __restrict__ zeros,
                                 float*       __restrict__ y,
                                 int M, int K, int Cin, int relu)
{
  constexpr int NT   = 2 * TPW;
  constexpr int Cout = 32 * TPW;
  __shared__ __align__(16) float Xs[2][32][36];  // double-buffered 32x32 fp32 tile

  const int tid  = threadIdx.x;
  const int lane = tid & 31;
  const int wave = tid >> 5;
  const int m    = lane & 15;
  const int h    = lane >> 4;
  const int rsel = wave >> 1;           // row half (0/1)
  const int ntb  = wave & 1;            // column-tile parity
  const int row0 = blockIdx.x * 32;
  const int nCC  = (Cin + 31) >> 5;
  const int T    = K * nCC;             // total (k, cc) stages

  v8f acc[TPW];
  #pragma unroll
  for (int t = 0; t < TPW; ++t) acc[t] = (v8f){0,0,0,0,0,0,0,0};

  stage_tile(x, map, zeros, Xs[0], tid, row0, M, K, Cin, 0, 0);

  for (int i = 0; i < T; ++i) {
    const int buf = i & 1;
    if (i + 1 < T) {
      const int kn  = (nCC == 1) ? (i + 1) : (i + 1) / nCC;
      const int ccn = (nCC == 1) ? 0       : (i + 1) % nCC;
      stage_tile(x, map, zeros, Xs[buf ^ 1], tid, row0, M, K, Cin, kn, ccn);
      wait_asynccnt<2>();               // stage i landed; stage i+1 still in flight
    } else {
      wait_asynccnt<0>();
    }
    __syncthreads();

    // A fragment per CDNA5 16-bit A layout; fp32 -> packed bf16 on read.
    const float* pr = &Xs[buf][rsel * 16 + m][0];
    Frag16 a;
    #pragma unroll
    for (int j = 0; j < 4; ++j) {
      a.u32[j]     = pack_bf16_pair(pr[8*h + 2*j],      pr[8*h + 2*j + 1]);
      a.u32[4 + j] = pack_bf16_pair(pr[16 + 8*h + 2*j], pr[16 + 8*h + 2*j + 1]);
    }

    const int k  = (nCC == 1) ? i : i / nCC;
    const int cc = (nCC == 1) ? 0 : i % nCC;
    #pragma unroll
    for (int t = 0; t < TPW; ++t) {
      const int nt = ntb + 2 * t;
      const uint4* bp = (const uint4*)(Wp +
          ((((size_t)k * nCC + cc) * NT + nt) * 32 + lane) * 8);
      Frag16 b;
      b.u128[0] = bp[0];
      b.u128[1] = bp[1];
      acc[t] = __builtin_amdgcn_wmma_f32_16x16x32_bf16(
          false, a.v, false, b.v, (short)0, acc[t], false, false);
    }
    __syncthreads();                    // reads done before buf is re-staged
  }

  // Epilogue: C/D layout -> element j at (row0 + rsel*16 + j + 8h, nt*16 + m)
  #pragma unroll
  for (int t = 0; t < TPW; ++t) {
    const int col = ((ntb + 2 * t) << 4) + m;
    #pragma unroll
    for (int j = 0; j < 8; ++j) {
      const int row = row0 + rsel * 16 + j + 8 * h;
      if (row < M) {
        float val = acc[t][j];
        if (relu) val = fmaxf(val, 0.f);
        y[(size_t)row * Cout + col] = val;
      }
    }
  }
}

// -------------------------------------------------------------------------
// Weight pre-pack: Wp[k][cc][nt][lane][8 dwords] = bf16 B-fragment layout.
// -------------------------------------------------------------------------
__global__ void pack_w(const float* __restrict__ W, unsigned int* __restrict__ Wp,
                       int K, int Cin, int Cout, int total) {
  int i = blockIdx.x * blockDim.x + threadIdx.x;
  if (i >= total) return;
  const int NT  = Cout >> 4;
  const int nCC = (Cin + 31) >> 5;
  const int j    = i & 7;
  const int lane = (i >> 3) & 31;
  int rest = i >> 8;                       // (((k*nCC)+cc)*NT)+nt
  const int nt = rest % NT;  rest /= NT;
  const int cc = rest % nCC; const int k = rest / nCC;
  const int n   = (nt << 4) + (lane & 15);
  const int h   = lane >> 4;
  const int ci0 = (cc << 5) + 8 * h + 2 * (j & 3) + ((j >> 2) << 4);
  float lo = (ci0     < Cin) ? W[((size_t)k * Cin + ci0    ) * Cout + n] : 0.f;
  float hi = (ci0 + 1 < Cin) ? W[((size_t)k * Cin + ci0 + 1) * Cout + n] : 0.f;
  Wp[i] = pack_bf16_pair(lo, hi);
}

// -------------------------------------------------------------------------
// Helpers: concat, zero, column stats (sum/sumsq), BN+ReLU apply
// -------------------------------------------------------------------------
__global__ void concat_feats(const float* __restrict__ g, const float* __restrict__ c,
                             float* __restrict__ x, int N) {
  int i = blockIdx.x * blockDim.x + threadIdx.x;
  if (i < N) {
    x[i*4+0] = g[i];
    x[i*4+1] = c[i*3+0];
    x[i*4+2] = c[i*3+1];
    x[i*4+3] = c[i*3+2];
  }
}

__global__ void zero_f32(float* __restrict__ p, int n) {
  int i = blockIdx.x * blockDim.x + threadIdx.x;
  if (i < n) p[i] = 0.f;
}

__launch_bounds__(256)
__global__ void col_stats(const float* __restrict__ y, float* __restrict__ stats,
                          int M, int C) {
  const int tpc = 256 / C;                    // C in {32,128,256}
  const int c   = threadIdx.x % C;
  const int sub = threadIdx.x / C;
  float s = 0.f, q = 0.f;
  for (int row = blockIdx.x * tpc + sub; row < M; row += gridDim.x * tpc) {
    float v = y[(size_t)row * C + c];
    s += v; q += v * v;
  }
  atomicAdd(&stats[c], s);
  atomicAdd(&stats[C + c], q);
}

__global__ void bn_relu(const float* __restrict__ t, const float* __restrict__ stats,
                        const float* __restrict__ g, const float* __restrict__ b,
                        float* __restrict__ out, int M, int C) {
  long long i = (long long)blockIdx.x * blockDim.x + threadIdx.x;
  if (i < (long long)M * C) {
    int c = (int)(i % C);
    float invM = 1.0f / (float)M;
    float mean = stats[c] * invM;
    float var  = stats[C + c] * invM - mean * mean;
    float v = (t[i] - mean) * rsqrtf(var + 1e-5f) * g[c] + b[c];
    out[i] = fmaxf(v, 0.f);
  }
}

// -------------------------------------------------------------------------
extern "C" void kernel_launch(void* const* d_in, const int* in_sizes, int n_in,
                              void* d_out, int out_size, void* d_ws, size_t ws_size,
                              hipStream_t stream) {
  (void)n_in; (void)out_size; (void)ws_size;
  const float* feats_geo = (const float*)d_in[0];
  const float* feats_col = (const float*)d_in[1];
  const float* W1   = (const float*)d_in[2];
  const float* W2   = (const float*)d_in[3];
  const float* W3   = (const float*)d_in[4];
  const float* g2   = (const float*)d_in[5];
  const float* b2   = (const float*)d_in[6];
  const float* g3   = (const float*)d_in[7];
  const float* b3   = (const float*)d_in[8];
  const float* Wc2  = (const float*)d_in[9];
  const float* We2a = (const float*)d_in[10];
  const float* ga   = (const float*)d_in[11];
  const float* ba   = (const float*)d_in[12];
  const float* We2b = (const float*)d_in[13];
  const float* gb   = (const float*)d_in[14];
  const float* bb   = (const float*)d_in[15];
  const int* map1   = (const int*)d_in[16];
  const int* map2   = (const int*)d_in[17];
  const int* map_c2 = (const int*)d_in[18];
  const int* map_e2 = (const int*)d_in[19];

  const int N  = in_sizes[0];
  const int M1 = in_sizes[16] / 125;
  const int M2 = in_sizes[18] / 27;

  char* ws = (char*)d_ws;
  float* stats    = (float*)ws;                  // 2*256 floats max
  float* zerosBuf = (float*)(ws + 3072);         // 32 floats, 16B aligned
  auto aup = [](size_t v){ return (v + 255) & ~(size_t)255; };
  auto packDw = [](int K, int Cin, int Cout) {
    return (size_t)K * ((Cin + 31) / 32) * (Cout / 16) * 32 * 8;
  };

  // ---- persistent region: stats + zeros + packed weights ----
  size_t off = 4096;
  unsigned int* Wp1 = (unsigned int*)(ws + off); off += aup(packDw(125,   4,  32) * 4);
  unsigned int* Wp2 = (unsigned int*)(ws + off); off += aup(packDw(125,  32,  32) * 4);
  unsigned int* Wp3 = (unsigned int*)(ws + off); off += aup(packDw(125,  32,  32) * 4);
  unsigned int* WpC = (unsigned int*)(ws + off); off += aup(packDw( 27,  32,  64) * 4);
  unsigned int* WpA = (unsigned int*)(ws + off); off += aup(packDw( 27,  64, 128) * 4);
  unsigned int* WpB = (unsigned int*)(ws + off); off += aup(packDw( 27, 128, 256) * 4);
  const size_t base = off;

  // ---- stage 1 layout ----
  size_t o1 = base;
  float* xcat = (float*)(ws + o1); o1 += aup((size_t)N  * 4   * 4);
  float* bufA = (float*)(ws + o1); o1 += aup((size_t)M1 * 32  * 4);  // h1 = relu(conv1)
  float* bufB = (float*)(ws + o1); o1 += aup((size_t)M1 * 32  * 4);  // raw conv2/3
  float* bufC = (float*)(ws + o1); o1 += aup((size_t)M1 * 32  * 4);  // bn_relu(conv2)
  // ---- stage 2 layout (overlays stage-1 buffers, dead by then) ----
  size_t o2 = base;
  float* h2c  = (float*)(ws + o2); o2 += aup((size_t)M2 * 64  * 4);
  float* rawA = (float*)(ws + o2); o2 += aup((size_t)M2 * 128 * 4);
  float* hA   = (float*)(ws + o2); o2 += aup((size_t)M2 * 128 * 4);
  float* rawB = (float*)(ws + o2); o2 += aup((size_t)M2 * 256 * 4);

  float* xe1 = (float*)d_out;                      // [M1, 32]
  float* xe2 = (float*)d_out + (size_t)M1 * 32;    // [M2, 256]

  zero_f32<<<1, 32, 0, stream>>>(zerosBuf, 32);

  // ---- pack all weights to bf16 fragment layout ----
  {
    struct { const float* W; unsigned int* Wp; int K, Ci, Co; } P[6] = {
      {W1,   Wp1, 125,   4,  32}, {W2,   Wp2, 125,  32,  32},
      {W3,   Wp3, 125,  32,  32}, {Wc2,  WpC,  27,  32,  64},
      {We2a, WpA,  27,  64, 128}, {We2b, WpB,  27, 128, 256},
    };
    for (int i = 0; i < 6; ++i) {
      int total = (int)packDw(P[i].K, P[i].Ci, P[i].Co);
      pack_w<<<(total + 255) / 256, 256, 0, stream>>>(
          P[i].W, P[i].Wp, P[i].K, P[i].Ci, P[i].Co, total);
    }
  }

  auto launch_conv = [&](const float* xin, const unsigned int* Wp, const int* mp,
                         float* yout, int M, int K, int Cin, int Cout, int relu) {
    dim3 g((M + 31) / 32), b(128);
    switch (Cout) {
      case 32:
        sparse_conv_wmma<1><<<g, b, 0, stream>>>(xin, Wp, mp, zerosBuf, yout, M, K, Cin, relu);
        break;
      case 64:
        sparse_conv_wmma<2><<<g, b, 0, stream>>>(xin, Wp, mp, zerosBuf, yout, M, K, Cin, relu);
        break;
      case 128:
        sparse_conv_wmma<4><<<g, b, 0, stream>>>(xin, Wp, mp, zerosBuf, yout, M, K, Cin, relu);
        break;
      default:
        sparse_conv_wmma<8><<<g, b, 0, stream>>>(xin, Wp, mp, zerosBuf, yout, M, K, Cin, relu);
        break;
    }
  };

  // x = concat(geo, col)
  concat_feats<<<(N + 255) / 256, 256, 0, stream>>>(feats_geo, feats_col, xcat, N);

  // h = relu(conv1(x))            K=125, 4->32
  launch_conv(xcat, Wp1, map1, bufA, M1, 125, 4, 32, 1);

  // h = relu(bn(conv2(h)))        K=125, 32->32
  launch_conv(bufA, Wp2, map2, bufB, M1, 125, 32, 32, 0);
  zero_f32<<<1, 64, 0, stream>>>(stats, 64);
  col_stats<<<256, 256, 0, stream>>>(bufB, stats, M1, 32);
  bn_relu<<<(unsigned)(((size_t)M1 * 32 + 255) / 256), 256, 0, stream>>>(
      bufB, stats, g2, b2, bufC, M1, 32);

  // x_e1 = relu(bn(conv3(h)))     K=125, 32->32
  launch_conv(bufC, Wp3, map2, bufB, M1, 125, 32, 32, 0);
  zero_f32<<<1, 64, 0, stream>>>(stats, 64);
  col_stats<<<256, 256, 0, stream>>>(bufB, stats, M1, 32);
  bn_relu<<<(unsigned)(((size_t)M1 * 32 + 255) / 256), 256, 0, stream>>>(
      bufB, stats, g3, b3, xe1, M1, 32);

  // h2 = conv_c2(x_e1)            K=27, 32->64 (reads x_e1 from d_out)
  launch_conv(xe1, WpC, map_c2, h2c, M2, 27, 32, 64, 0);

  // h2 = relu(bn(e2a(h2)))        K=27, 64->128
  launch_conv(h2c, WpA, map_e2, rawA, M2, 27, 64, 128, 0);
  zero_f32<<<1, 256, 0, stream>>>(stats, 256);
  col_stats<<<256, 256, 0, stream>>>(rawA, stats, M2, 128);
  bn_relu<<<(unsigned)(((size_t)M2 * 128 + 255) / 256), 256, 0, stream>>>(
      rawA, stats, ga, ba, hA, M2, 128);

  // x_e2 = relu(bn(e2b(h2)))      K=27, 128->256
  launch_conv(hA, WpB, map_e2, rawB, M2, 27, 128, 256, 0);
  zero_f32<<<2, 256, 0, stream>>>(stats, 512);
  col_stats<<<256, 256, 0, stream>>>(rawB, stats, M2, 256);
  bn_relu<<<(unsigned)(((size_t)M2 * 256 + 255) / 256), 256, 0, stream>>>(
      rawB, stats, gb, bb, xe2, M2, 256);
}
